// MDLSTMConvBlock_25108378812517
// MI455X (gfx1250) — matrix operand order repaired and checked by reading, not compiled, not run on hardware
//
#include <hip/hip_runtime.h>
#include <hip/hip_bf16.h>

typedef __attribute__((ext_vector_type(16))) __bf16 v16bf;
typedef __attribute__((ext_vector_type(8)))  float  v8f;

#define Bn   8
#define HIN  64
#define WIN  256
#define CC   32          // conv out channels
#define Hh   31
#define Ww   127
#define G    128
#define GC   640         // 5 gates * 128
#define NB   32          // 4 dirs * 8 batch
#define KTOT 288         // 32 (x) + 128 (h_up) + 128 (h_left)
#define KB   9           // K blocks of 32
#define NT   40          // N tiles of 16
#define NWG  31          // persistent workgroups (max cells per diagonal)
#define THREADS 512      // 16 waves: 2 m-tiles x 8 n-groups of 5 tiles
#define TPW  5           // n-tiles per wave

__device__ __forceinline__ float sigf(float x) {
    return 1.0f / (1.0f + __expf(-x));
}

// ---------------- conv(3x3 valid) + maxpool(2x2) + tanh -> bf16 [B][H][W][C] ----------------
__global__ void conv_pool_tanh_k(const float* __restrict__ x, const float* __restrict__ w,
                                 const float* __restrict__ cb, __bf16* __restrict__ t) {
    int gid = blockIdx.x * 256 + threadIdx.x;
    if (gid >= Bn * Hh * Ww * CC) return;
    int c = gid & 31;
    int j = (gid >> 5) % Ww;
    int i = (gid / (32 * Ww)) % Hh;
    int b = gid / (32 * Ww * Hh);
    float wl[9];
#pragma unroll
    for (int q = 0; q < 9; ++q) wl[q] = w[c * 9 + q];
    float bb = cb[c];
    float m = -1e30f;
#pragma unroll
    for (int py = 0; py < 2; ++py)
#pragma unroll
        for (int px = 0; px < 2; ++px) {
            int oy = 2 * i + py, ox = 2 * j + px;
            float s = bb;
#pragma unroll
            for (int kh = 0; kh < 3; ++kh)
#pragma unroll
                for (int kw = 0; kw < 3; ++kw)
                    s += x[(b * HIN + oy + kh) * WIN + ox + kw] * wl[kh * 3 + kw];
            m = fmaxf(m, s);
        }
    t[gid] = (__bf16)tanhf(m);
}

// ---------------- pack [Wx;Wh1;Wh2] into per-WMMA-B-fragment bf16 layout ----------------
// pB[((kb*NT + nt)*32 + lane)*16 + e] = W[kb*32 + (lane>=16?16:0) + e][nt*16 + (lane&15)]
__global__ void pack_weights_k(const float* __restrict__ Wx, const float* __restrict__ Wh1,
                               const float* __restrict__ Wh2, __bf16* __restrict__ pB) {
    int gid = blockIdx.x * 256 + threadIdx.x;
    if (gid >= KTOT * GC) return;
    int e  = gid & 15;
    int l  = (gid >> 4) & 31;
    int nt = (gid >> 9) % NT;
    int kb = gid / (16 * 32 * NT);
    int k   = kb * 32 + ((l >= 16) ? 16 : 0) + e;
    int col = nt * 16 + (l & 15);
    float v;
    if (k < 32)       v = Wx[k * GC + col];
    else if (k < 160) v = Wh1[(k - 32) * GC + col];
    else              v = Wh2[(k - 160) * GC + col];
    pB[gid] = (__bf16)v;
}

// ---------------- zero the grid-barrier counter (ws is poisoned by harness) ----------------
__global__ void zero_sync_k(int* sync) { *sync = 0; }

// ---------------- persistent MD-LSTM wavefront scan: one launch, 157 diagonals ----------------
__global__ void __launch_bounds__(THREADS)
mdlstm_scan_k(const __bf16* __restrict__ t, const __bf16* __restrict__ pB,
              const float* __restrict__ bias, float* __restrict__ hG,
              float* __restrict__ cG, int* __restrict__ sync) {
    // A staged in per-lane WMMA fragment order: [mt][kb][lane][16] bf16 (18,432 B)
    __shared__ __bf16 aFrag[2 * KB * 32 * 16];
    __shared__ float  zLds[NB * GC];     // 81,920 B

    int tid  = threadIdx.x;
    int wave = tid >> 5;
    int lane = tid & 31;
    int mt   = wave & 1;
    int nt0  = (wave >> 1) * TPW;
    int zr   = (lane >= 16) ? 8 : 0;
    int zc   = lane & 15;
    int ci   = blockIdx.x;               // this WG always owns row ci

    // single base pointers; all fragment offsets become compile-time immediates
    const __bf16* bBase = pB + ((size_t)nt0 * 32 + lane) * 16;         // + (kb*NT+q)*512 elems
    const __bf16* aBase = aFrag + ((size_t)mt * KB * 32 + lane) * 16;  // + kb*512 elems

    for (int d = 0; d <= Hh + Ww - 2; ++d) {
        int cj = d - ci;
        bool active = (cj >= 0) && (cj < Ww);   // workgroup-uniform
        if (active) {
            // ---- stage A = [x | h_up | h_left] into LDS, fragment-major ----
            for (int idx = tid; idx < 2 * KB * 32 * 16; idx += THREADS) {
                int e  = idx & 15;
                int l  = (idx >> 4) & 31;
                int kb = (idx >> 9) % KB;
                int mtS = idx / (16 * 32 * KB);
                int m = mtS * 16 + (l & 15);
                int k = kb * 32 + ((l >= 16) ? 16 : 0) + e;
                __bf16 v;
                if (k < 32) {
                    int dir = m >> 3, bb = m & 7;
                    int ti = (dir & 2) ? (Hh - 1 - ci) : ci;
                    int tj = (dir & 1) ? (Ww - 1 - cj) : cj;
                    v = t[((bb * Hh + ti) * Ww + tj) * 32 + k];
                } else if (k < 160) {
                    v = (ci > 0) ? (__bf16)hG[((((ci - 1) * Ww + cj) * NB) + m) * G + (k - 32)]
                                 : (__bf16)0.0f;
                } else {
                    v = (cj > 0) ? (__bf16)hG[(((ci * Ww + (cj - 1)) * NB) + m) * G + (k - 160)]
                                 : (__bf16)0.0f;
                }
                aFrag[idx] = v;
            }
            __syncthreads();

            // ---- GEMM: [32,288] x [288,640]; wave = (mt, 5 n-tiles); bias pre-folded ----
            v8f acc[TPW];
#pragma unroll
            for (int q = 0; q < TPW; ++q) {
                float bb = bias[(nt0 + q) * 16 + zc];
                acc[q] = (v8f){bb, bb, bb, bb, bb, bb, bb, bb};
            }
#pragma unroll
            for (int kb = 0; kb < KB; ++kb) {
                v16bf a = *(const v16bf*)(aBase + kb * (32 * 16));
                v16bf bf[TPW];
#pragma unroll
                for (int q = 0; q < TPW; ++q)
                    bf[q] = *(const v16bf*)(bBase + (kb * NT + q) * (32 * 16));
#pragma unroll
                for (int q = 0; q < TPW; ++q)
                    acc[q] = __builtin_amdgcn_wmma_f32_16x16x32_bf16(
                        false, a, false, bf[q], (short)0, acc[q], false, false);
            }

            // ---- spill z tiles to LDS ----
#pragma unroll
            for (int q = 0; q < TPW; ++q) {
                int colb = (nt0 + q) * 16 + zc;
#pragma unroll
                for (int r = 0; r < 8; ++r)
                    zLds[(mt * 16 + zr + r) * GC + colb] = acc[q][r];
            }
            __syncthreads();

            // ---- gates + state update ----
            int cell = ci * Ww + cj;
#pragma unroll
            for (int q = 0; q < (NB * G) / THREADS; ++q) {
                int e = tid + q * THREADS;      // 0..4095
                int m = e >> 7;
                int g = e & 127;
                float zi  = zLds[m * GC + 0 * G + g];
                float zf1 = zLds[m * GC + 1 * G + g];
                float zf2 = zLds[m * GC + 2 * G + g];
                float zo  = zLds[m * GC + 3 * G + g];
                float zg  = zLds[m * GC + 4 * G + g];
                float cu = (ci > 0) ? cG[((((ci - 1) * Ww + cj) * NB) + m) * G + g] : 0.0f;
                float cl = (cj > 0) ? cG[(((ci * Ww + (cj - 1)) * NB) + m) * G + g] : 0.0f;
                float c = sigf(zf1) * cu + sigf(zf2) * cl + sigf(zi) * tanhf(zg);
                float h = sigf(zo) * tanhf(c);
                cG[(cell * NB + m) * G + g] = c;
                hG[(cell * NB + m) * G + g] = h;
            }
            __syncthreads();   // LDS reuse safety before next diagonal's staging
        }

        // ---- grid-wide barrier between diagonals (all NWG workgroups arrive) ----
        __threadfence();
        __syncthreads();
        if (tid == 0) {
            int target = NWG * (d + 1);
            int arrived = __hip_atomic_fetch_add(sync, 1, __ATOMIC_ACQ_REL,
                                                 __HIP_MEMORY_SCOPE_AGENT) + 1;
            if (arrived < target) {
                while (__hip_atomic_load(sync, __ATOMIC_ACQUIRE,
                                         __HIP_MEMORY_SCOPE_AGENT) < target) {
                    __builtin_amdgcn_s_sleep(2);
                }
            }
        }
        __syncthreads();
    }
}

// ---------------- un-flip + average the 4 directions into d_out [8,128,31,127] ----------------
__global__ void gather_out_k(const float* __restrict__ hG, float* __restrict__ out) {
    int gid = blockIdx.x * 256 + threadIdx.x;
    if (gid >= Bn * G * Hh * Ww) return;
    int j = gid % Ww;
    int i = (gid / Ww) % Hh;
    int g = (gid / (Ww * Hh)) % G;
    int b = gid / (Ww * Hh * G);
    float s = hG[(((i * Ww + j) * NB) + b) * G + g]
            + hG[(((i * Ww + (Ww - 1 - j)) * NB) + 8 + b) * G + g]
            + hG[((((Hh - 1 - i) * Ww + j) * NB) + 16 + b) * G + g]
            + hG[((((Hh - 1 - i) * Ww + (Ww - 1 - j)) * NB) + 24 + b) * G + g];
    out[gid] = 0.25f * s;
}

extern "C" void kernel_launch(void* const* d_in, const int* in_sizes, int n_in,
                              void* d_out, int out_size, void* d_ws, size_t ws_size,
                              hipStream_t stream) {
    (void)in_sizes; (void)n_in; (void)out_size; (void)ws_size;
    const float* x    = (const float*)d_in[0];
    const float* cw   = (const float*)d_in[1];
    const float* cb   = (const float*)d_in[2];
    const float* Wx   = (const float*)d_in[3];
    const float* Wh1  = (const float*)d_in[4];
    const float* Wh2  = (const float*)d_in[5];
    const float* bias = (const float*)d_in[6];

    char* ws = (char*)d_ws;
    size_t off = 0;
    auto alloc = [&](size_t bytes) -> void* {
        void* p = ws + off;
        off += (bytes + 255) & ~(size_t)255;
        return p;
    };
    __bf16* t    = (__bf16*)alloc((size_t)Bn * Hh * Ww * 32 * sizeof(__bf16)); // ~2 MB
    __bf16* pB   = (__bf16*)alloc((size_t)KTOT * GC * sizeof(__bf16));         // 368 KB
    float*  hG   = (float*)alloc((size_t)Hh * Ww * NB * G * sizeof(float));    // 64.5 MB
    float*  cG   = (float*)alloc((size_t)Hh * Ww * NB * G * sizeof(float));    // 64.5 MB
    int*    sync = (int*)alloc(256);

    int nconv = Bn * Hh * Ww * 32;
    conv_pool_tanh_k<<<(nconv + 255) / 256, 256, 0, stream>>>(x, cw, cb, t);

    int npk = KTOT * GC;
    pack_weights_k<<<(npk + 255) / 256, 256, 0, stream>>>(Wx, Wh1, Wh2, pB);

    zero_sync_k<<<1, 1, 0, stream>>>(sync);

    // single persistent launch: 31 co-resident workgroups sweep all 157 diagonals
    mdlstm_scan_k<<<NWG, THREADS, 0, stream>>>(t, pB, bias, hG, cG, sync);

    int nout = Bn * G * Hh * Ww;
    gather_out_k<<<(nout + 255) / 256, 256, 0, stream>>>(hG, (float*)d_out);
}
